// GAT_27960237097246
// MI455X (gfx1250) — matrix operand order repaired and checked by reading, not compiled, not run on hardware
//
#include <hip/hip_runtime.h>
#include <cstdint>

// ---------------------------------------------------------------------------
// GAT on MI455X (gfx1250): f16 WMMA GEMMs (f32 accumulate), f16 edge
// intermediates, algebraically fused attention projections, fused edge update.
// GEMM waves block 4 N-tiles to reuse each A fragment across 4 WMMAs.
// ---------------------------------------------------------------------------

typedef __attribute__((ext_vector_type(16))) _Float16 v16h;
typedef __attribute__((ext_vector_type(8)))  float    v8f;

#define W_HID 512

union HV { v16h v; _Float16 h[16]; };

// -------------------- WMMA A-fragment loader (16x32 f16) -------------------
// ISA layout: lanes 0-15 -> M=lane, halves e0..7 = K[kb..kb+7], e8..15 = K[kb+16..kb+23]
//             lanes 16-31 -> M=lane-16, e0..7 = K[kb+8..kb+15], e8..15 = K[kb+24..kb+31]
template<typename T, bool TRANS>
__device__ __forceinline__ v16h load_a_frag(const T* __restrict__ A, int lda,
                                            int row, int kb, int lane) {
  HV u;
  const int base = kb + ((lane & 16) ? 8 : 0);
  if constexpr (!TRANS) {
    const T* p = A + (size_t)row * lda + base;
#pragma unroll
    for (int e = 0; e < 8; ++e) u.h[e]     = (_Float16)p[e];
#pragma unroll
    for (int e = 0; e < 8; ++e) u.h[8 + e] = (_Float16)p[16 + e];
  } else {
    // A'[m][k] = src[k*lda + m]
    const T* p = A + (size_t)base * lda + row;
#pragma unroll
    for (int e = 0; e < 8; ++e) u.h[e]     = (_Float16)p[(size_t)e * lda];
#pragma unroll
    for (int e = 0; e < 8; ++e) u.h[8 + e] = (_Float16)p[(size_t)(16 + e) * lda];
  }
  return u.v;
}

// -------------------- generic WMMA GEMM ------------------------------------
// Wave computes a 16 x (16*NB) strip: one A fragment feeds NB WMMAs per K-step.
// C(f32/f16) = A(MxK) * Bpacked(KxNpad) + bias ; optional f32 and f16 outputs.
template<typename AT, bool ATRANS, int NB>
__global__ __launch_bounds__(256)
void k_gemm_wmma(const AT* __restrict__ A, int lda,
                 const _Float16* __restrict__ Bp, int K, int Npad,
                 const float* __restrict__ bias,
                 float* __restrict__ Cf, int ldcf, int Nout,
                 _Float16* __restrict__ Ch, int ldch,
                 int M)
{
  const int lane = threadIdx.x & 31;
  const int w    = (blockIdx.x * blockDim.x + threadIdx.x) >> 5;
  const int Ng   = (Npad >> 4) / NB;         // groups of NB N-tiles
  const int Mt   = M >> 4;
  if (w >= Mt * Ng) return;                  // wave-uniform exit
  const int tg = w % Ng, tm = w / Ng;
  const int tn0 = tg * NB;
  const int Kt = K >> 5;

  v8f acc[NB];
#pragma unroll
  for (int t = 0; t < NB; ++t) acc[t] = (v8f){};

  const int row = tm * 16 + (lane & 15);
  // B tile (tn, kt) lives at ((tn*Kt + kt)*32 + lane)*16 halves
  const _Float16* bbase = Bp + (size_t)tn0 * Kt * 512 + (size_t)lane * 16;
  for (int kt = 0; kt < Kt; ++kt) {
    v16h a = load_a_frag<AT, ATRANS>(A, lda, row, kt * 32, lane);
    const _Float16* bk = bbase + (size_t)kt * 512;
#pragma unroll
    for (int t = 0; t < NB; ++t) {
      v16h b = *reinterpret_cast<const v16h*>(bk + (size_t)t * Kt * 512);
      acc[t] = __builtin_amdgcn_wmma_f32_16x16x32_f16(false, a, false, b,
                                                      (short)0, acc[t], false, false);
    }
  }

  // C/D layout: lane -> col = (lane&15); VGPR r -> m = r + 8*(lane>=16)
  const int mbase = tm * 16 + ((lane & 16) ? 8 : 0);
#pragma unroll
  for (int t = 0; t < NB; ++t) {
    const int col = (tn0 + t) * 16 + (lane & 15);
    const float bc = bias ? bias[col] : 0.0f;
#pragma unroll
    for (int r = 0; r < 8; ++r) {
      float v = acc[t][r] + bc;
      if (Cf && col < Nout) Cf[(size_t)(mbase + r) * ldcf + col] = v;
      if (Ch)               Ch[(size_t)(mbase + r) * ldch + col] = (_Float16)v;
    }
  }
}

// -------------------- pack f32 KxN weight into WMMA B layout ---------------
// B layout: lanes 0-15 hold K=kb..kb+7, kb+16..kb+23; lanes 16-31 hold K=kb+8..kb+15, kb+24..kb+31.
__global__ void k_pack_b(const float* __restrict__ Wsrc, _Float16* __restrict__ Bp,
                         int K, int N, int Npad)
{
  int idx = blockIdx.x * blockDim.x + threadIdx.x;
  if (idx >= Npad * K) return;
  int e    = idx & 15;
  int lane = (idx >> 4) & 31;
  int rest = idx >> 9;
  int Kt   = K >> 5;
  int kt   = rest % Kt;
  int nt   = rest / Kt;
  int k = kt * 32 + ((lane & 16) ? 8 : 0) + (e & 7) + ((e & 8) ? 16 : 0);
  int nn = nt * 16 + (lane & 15);
  float v = (nn < N) ? Wsrc[(size_t)k * N + nn] : 0.0f;
  Bp[idx] = (_Float16)v;
}

// -------------------- small: C[K][8] = Wsrc(KxMw) @ Wa(Mwx8) ---------------
__global__ void k_small_AB(const float* __restrict__ Wsrc, const float* __restrict__ Wa,
                           float* __restrict__ C, int K, int Mw)
{
  int idx = blockIdx.x * blockDim.x + threadIdx.x;
  if (idx >= K * 8) return;
  int k = idx >> 3, h = idx & 7;
  float s = 0.f;
  for (int m = 0; m < Mw; ++m) s += Wsrc[(size_t)k * Mw + m] * Wa[m * 8 + h];
  C[idx] = s;
}

// out[0..15] = (h<8) ? b @ Wa[:,h] : 0   (bias projected through Wa, padded)
__global__ void k_bias_proj(const float* __restrict__ bvec, const float* __restrict__ Wa,
                            float* __restrict__ outp, int Mw)
{
  int h = threadIdx.x;
  if (h >= 16) return;
  float s = 0.f;
  if (h < 8) for (int m = 0; m < Mw; ++m) s += bvec[m] * Wa[m * 8 + h];
  outp[h] = s;
}

// -------------------- attention scores + row softmax -----------------------
// block = (b*n + i), threads j = 0..n-1
__global__ __launch_bounds__(256)
void k_scores_softmax(const float* __restrict__ nodes, int en,
                      const float* __restrict__ q, const float* __restrict__ qe,
                      float* __restrict__ an, float* __restrict__ ae, int n)
{
  const int j  = threadIdx.x;
  const int bi = blockIdx.x;                     // b*n + i
  __shared__ float qs[8];
  __shared__ float red[256];
  if (j < 8) qs[j] = q[(size_t)bi * 8 + j];
  __syncthreads();
  const float mk = (nodes[(size_t)bi * en + (en - 1)] == 0.0f) ? 1.0f : 0.0f;
  const int b0 = bi / n;

  // ---- node scores: sn[j] = <q_i, q_j>/8 * mask_i ----
  float sn = 0.f;
  {
    const float* qj = q + ((size_t)b0 * n + j) * 8;
#pragma unroll
    for (int h = 0; h < 8; ++h) sn += qs[h] * qj[h];
  }
  sn *= 0.125f * mk;
  red[j] = sn; __syncthreads();
  for (int s = 128; s > 0; s >>= 1) { if (j < s) red[j] = fmaxf(red[j], red[j + s]); __syncthreads(); }
  float mx = red[0]; __syncthreads();
  float ex = __expf(sn - mx);
  red[j] = ex; __syncthreads();
  for (int s = 128; s > 0; s >>= 1) { if (j < s) red[j] += red[j + s]; __syncthreads(); }
  an[(size_t)bi * n + j] = ex / red[0];
  __syncthreads();

  // ---- edge scores: se[j] = <q_i, qe_ij>/8 * mask_i ----
  float se = 0.f;
  {
    const float* qj = qe + ((size_t)bi * n + j) * 8;
#pragma unroll
    for (int h = 0; h < 8; ++h) se += qs[h] * qj[h];
  }
  se *= 0.125f * mk;
  red[j] = se; __syncthreads();
  for (int s = 128; s > 0; s >>= 1) { if (j < s) red[j] = fmaxf(red[j], red[j + s]); __syncthreads(); }
  mx = red[0]; __syncthreads();
  ex = __expf(se - mx);
  red[j] = ex; __syncthreads();
  for (int s = 128; s > 0; s >>= 1) { if (j < s) red[j] += red[j + s]; __syncthreads(); }
  ae[(size_t)bi * n + j] = ex / red[0];
}

// -------------------- fused edge update + agg_e ----------------------------
// In one pass over ef[b,i,:,:] (f16, in place):
//   agg_e[b,i,k]  = sum_j ae * ef
//   ef            = elu(ef*(1+ae) + an*nf[b,i,k])   (becomes h_e)
__global__ __launch_bounds__(256)
void k_edge_update(_Float16* __restrict__ ef,
                   const float* __restrict__ an, const float* __restrict__ ae,
                   const float* __restrict__ nf, float* __restrict__ agg_e, int n)
{
  const int row = blockIdx.x;                    // b*n + i
  const int t   = threadIdx.x;                   // owns k = t and k = t+256
  const float nf0 = nf[(size_t)row * W_HID + t];
  const float nf1 = nf[(size_t)row * W_HID + t + 256];
  float agg0 = 0.f, agg1 = 0.f;
  const size_t ebase = (size_t)row * n * W_HID;
  for (int j = 0; j < n; ++j) {
    const float aej = ae[(size_t)row * n + j];
    const float anj = an[(size_t)row * n + j];
    _Float16* ep = ef + ebase + (size_t)j * W_HID;
    float e0 = (float)ep[t];
    float e1 = (float)ep[t + 256];
    agg0 += aej * e0;
    agg1 += aej * e1;
    float u0 = e0 * (1.f + aej) + anj * nf0;
    float u1 = e1 * (1.f + aej) + anj * nf1;
    ep[t]       = (_Float16)(u0 > 0.f ? u0 : __expf(u0) - 1.f);
    ep[t + 256] = (_Float16)(u1 > 0.f ? u1 : __expf(u1) - 1.f);
  }
  agg_e[(size_t)row * W_HID + t]       = agg0;
  agg_e[(size_t)row * W_HID + t + 256] = agg1;
}

// -------------------- node update: h_n = elu(nf + awn + agg_e) -------------
__global__ void k_node_update(const float* __restrict__ nf, const float* __restrict__ awn,
                              const float* __restrict__ agg, float* __restrict__ hn, int total)
{
  int i = blockIdx.x * blockDim.x + threadIdx.x;
  if (i >= total) return;
  float u = nf[i] + awn[i] + agg[i];
  hn[i] = u > 0.f ? u : __expf(u) - 1.f;
}

// ===========================================================================
extern "C" void kernel_launch(void* const* d_in, const int* in_sizes, int n_in,
                              void* d_out, int out_size, void* d_ws, size_t ws_size,
                              hipStream_t stream)
{
  (void)in_sizes; (void)n_in; (void)out_size; (void)ws_size;
  const int b = 2, n = 256, en = 128, ee = 64, W = 512;
  const int BN  = b * n;        // 512
  const int BNN = b * n * n;    // 131072

  const float* nodes = (const float*)d_in[0];
  const float* edges = (const float*)d_in[1];
  const float* Wn0 = (const float*)d_in[2];
  const float* bn0 = (const float*)d_in[3];
  const float* We0 = (const float*)d_in[4];
  const float* be0 = (const float*)d_in[5];
  const float* Wa0 = (const float*)d_in[6];
  const float* Wn1 = (const float*)d_in[7];
  const float* bn1 = (const float*)d_in[8];
  const float* We1 = (const float*)d_in[9];
  const float* be1 = (const float*)d_in[10];
  const float* Wa1 = (const float*)d_in[11];
  const float* Wfn = (const float*)d_in[12];
  const float* bfn = (const float*)d_in[13];
  const float* Wfe = (const float*)d_in[14];
  const float* bfe = (const float*)d_in[15];

  char* wsb = (char*)d_ws;
  size_t off = 0;
  auto alloc = [&](size_t bytes) -> void* {
    void* p = wsb + off;
    off += (bytes + 255) & ~(size_t)255;
    return p;
  };

  // packed weights (f16, WMMA-B layout)
  _Float16* pWn0 = (_Float16*)alloc((size_t)en * W * 2);
  _Float16* pWe0 = (_Float16*)alloc((size_t)ee * W * 2);
  _Float16* pWn1 = (_Float16*)alloc((size_t)W * W * 2);
  _Float16* pWe1 = (_Float16*)alloc((size_t)W * W * 2);
  _Float16* pWfn = (_Float16*)alloc((size_t)W * en * 2);
  _Float16* pWfe = (_Float16*)alloc((size_t)W * ee * 2);
  // combined attention projections C = Wx @ Wa  (K x 8) + packed versions
  float* Cn0 = (float*)alloc((size_t)en * 8 * 4);
  float* Ce0 = (float*)alloc((size_t)ee * 8 * 4);
  float* Cn1 = (float*)alloc((size_t)W * 8 * 4);
  float* Ce1 = (float*)alloc((size_t)W * 8 * 4);
  _Float16* pCn0 = (_Float16*)alloc((size_t)en * 16 * 2);
  _Float16* pCe0 = (_Float16*)alloc((size_t)ee * 16 * 2);
  _Float16* pCn1 = (_Float16*)alloc((size_t)W * 16 * 2);
  _Float16* pCe1 = (_Float16*)alloc((size_t)W * 16 * 2);
  float* bqn0 = (float*)alloc(16 * 4);
  float* bqe0 = (float*)alloc(16 * 4);
  float* bqn1 = (float*)alloc(16 * 4);
  float* bqe1 = (float*)alloc(16 * 4);
  // per-layer working buffers
  float* nf   = (float*)alloc((size_t)BN * W * 4);
  float* q    = (float*)alloc((size_t)BN * 8 * 4);
  float* qe   = (float*)alloc((size_t)BNN * 8 * 4);
  float* an   = (float*)alloc((size_t)BNN * 4);
  float* ae   = (float*)alloc((size_t)BNN * 4);
  float* awn  = (float*)alloc((size_t)BN * W * 4);
  float* agg  = (float*)alloc((size_t)BN * W * 4);
  float* hn0  = (float*)alloc((size_t)BN * W * 4);
  float* hn1  = (float*)alloc((size_t)BN * W * 4);
  _Float16* nfB = (_Float16*)alloc((size_t)b * n * W * 2);   // nf packed per batch
  // big edge hidden states (f16)
  _Float16* efA = (_Float16*)alloc((size_t)BNN * W * 2);
  _Float16* efB = (_Float16*)alloc((size_t)BNN * W * 2);

  auto gb = [](int M, int Npad, int NB) {
    long waves = (long)(M / 16) * ((Npad / 16) / NB);
    return (int)((waves * 32 + 255) / 256);
  };
  auto eb = [](int total) { return (total + 255) / 256; };

  // ---------------- setup: pack weights, combined projections --------------
  k_pack_b<<<eb(en * W), 256, 0, stream>>>(Wn0, pWn0, en, W, W);
  k_pack_b<<<eb(ee * W), 256, 0, stream>>>(We0, pWe0, ee, W, W);
  k_pack_b<<<eb(W * W), 256, 0, stream>>>(Wn1, pWn1, W, W, W);
  k_pack_b<<<eb(W * W), 256, 0, stream>>>(We1, pWe1, W, W, W);
  k_pack_b<<<eb(W * en), 256, 0, stream>>>(Wfn, pWfn, W, en, en);
  k_pack_b<<<eb(W * ee), 256, 0, stream>>>(Wfe, pWfe, W, ee, ee);
  k_small_AB<<<eb(en * 8), 256, 0, stream>>>(Wn0, Wa0, Cn0, en, W);
  k_small_AB<<<eb(ee * 8), 256, 0, stream>>>(We0, Wa0, Ce0, ee, W);
  k_small_AB<<<eb(W * 8), 256, 0, stream>>>(Wn1, Wa1, Cn1, W, W);
  k_small_AB<<<eb(W * 8), 256, 0, stream>>>(We1, Wa1, Ce1, W, W);
  k_pack_b<<<eb(en * 16), 256, 0, stream>>>(Cn0, pCn0, en, 8, 16);
  k_pack_b<<<eb(ee * 16), 256, 0, stream>>>(Ce0, pCe0, ee, 8, 16);
  k_pack_b<<<eb(W * 16), 256, 0, stream>>>(Cn1, pCn1, W, 8, 16);
  k_pack_b<<<eb(W * 16), 256, 0, stream>>>(Ce1, pCe1, W, 8, 16);
  k_bias_proj<<<1, 16, 0, stream>>>(bn0, Wa0, bqn0, W);
  k_bias_proj<<<1, 16, 0, stream>>>(be0, Wa0, bqe0, W);
  k_bias_proj<<<1, 16, 0, stream>>>(bn1, Wa1, bqn1, W);
  k_bias_proj<<<1, 16, 0, stream>>>(be1, Wa1, bqe1, W);

  // ---------------- layer 0 ------------------------------------------------
  // nf = nodes @ Wn0 + bn0 ; q = nodes @ (Wn0@Wa0) + bn0@Wa0
  k_gemm_wmma<float, false, 4><<<gb(BN, W, 4), 256, 0, stream>>>(
      nodes, en, pWn0, en, W, bn0, nf, W, W, nullptr, 0, BN);
  k_gemm_wmma<float, false, 1><<<gb(BN, 16, 1), 256, 0, stream>>>(
      nodes, en, pCn0, en, 16, bqn0, q, 8, 8, nullptr, 0, BN);
  // ef0 (f16) ; qe0 = edges @ (We0@Wa0) + be0@Wa0
  k_gemm_wmma<float, false, 4><<<gb(BNN, W, 4), 256, 0, stream>>>(
      edges, ee, pWe0, ee, W, be0, nullptr, 0, 0, efA, W, BNN);
  k_gemm_wmma<float, false, 1><<<gb(BNN, 16, 1), 256, 0, stream>>>(
      edges, ee, pCe0, ee, 16, bqe0, qe, 8, 8, nullptr, 0, BNN);
  k_scores_softmax<<<BN, 256, 0, stream>>>(nodes, en, q, qe, an, ae, n);
  // awn = an^T @ nf (per batch)
  k_pack_b<<<eb(W * n), 256, 0, stream>>>(nf, nfB, n, W, W);
  k_pack_b<<<eb(W * n), 256, 0, stream>>>(nf + (size_t)n * W, nfB + (size_t)n * W, n, W, W);
  k_gemm_wmma<float, true, 4><<<gb(n, W, 4), 256, 0, stream>>>(
      an, n, nfB, n, W, nullptr, awn, W, W, nullptr, 0, n);
  k_gemm_wmma<float, true, 4><<<gb(n, W, 4), 256, 0, stream>>>(
      an + (size_t)n * n, n, nfB + (size_t)n * W, n, W, nullptr,
      awn + (size_t)n * W, W, W, nullptr, 0, n);
  // fused edge update (efA -> h_e0 in place) + agg_e ; then node update
  k_edge_update<<<BN, 256, 0, stream>>>(efA, an, ae, nf, agg, n);
  k_node_update<<<eb(BN * W), 256, 0, stream>>>(nf, awn, agg, hn0, BN * W);

  // ---------------- layer 1 ------------------------------------------------
  k_gemm_wmma<float, false, 4><<<gb(BN, W, 4), 256, 0, stream>>>(
      hn0, W, pWn1, W, W, bn1, nf, W, W, nullptr, 0, BN);
  k_gemm_wmma<float, false, 1><<<gb(BN, 16, 1), 256, 0, stream>>>(
      hn0, W, pCn1, W, 16, bqn1, q, 8, 8, nullptr, 0, BN);
  k_gemm_wmma<_Float16, false, 4><<<gb(BNN, W, 4), 256, 0, stream>>>(
      efA, W, pWe1, W, W, be1, nullptr, 0, 0, efB, W, BNN);
  k_gemm_wmma<_Float16, false, 1><<<gb(BNN, 16, 1), 256, 0, stream>>>(
      efA, W, pCe1, W, 16, bqe1, qe, 8, 8, nullptr, 0, BNN);
  k_scores_softmax<<<BN, 256, 0, stream>>>(nodes, en, q, qe, an, ae, n);
  k_pack_b<<<eb(W * n), 256, 0, stream>>>(nf, nfB, n, W, W);
  k_pack_b<<<eb(W * n), 256, 0, stream>>>(nf + (size_t)n * W, nfB + (size_t)n * W, n, W, W);
  k_gemm_wmma<float, true, 4><<<gb(n, W, 4), 256, 0, stream>>>(
      an, n, nfB, n, W, nullptr, awn, W, W, nullptr, 0, n);
  k_gemm_wmma<float, true, 4><<<gb(n, W, 4), 256, 0, stream>>>(
      an + (size_t)n * n, n, nfB + (size_t)n * W, n, W, nullptr,
      awn + (size_t)n * W, W, W, nullptr, 0, n);
  k_edge_update<<<BN, 256, 0, stream>>>(efB, an, ae, nf, agg, n);
  k_node_update<<<eb(BN * W), 256, 0, stream>>>(nf, awn, agg, hn1, BN * W);

  // ---------------- final projections --------------------------------------
  float* out_nodes = (float*)d_out;
  float* out_edges = out_nodes + (size_t)BN * en;
  k_gemm_wmma<float, false, 4><<<gb(BN, en, 4), 256, 0, stream>>>(
      hn1, W, pWfn, W, en, bfn, out_nodes, en, en, nullptr, 0, BN);
  k_gemm_wmma<_Float16, false, 4><<<gb(BNN, ee, 4), 256, 0, stream>>>(
      efB, W, pWfe, W, ee, bfe, out_edges, ee, ee, nullptr, 0, BNN);
}
